// Decoder_137438953915
// MI455X (gfx1250) — compile-verified, matrix operand order checked by reading, not run on hardware
//
#include <hip/hip_runtime.h>
#include <hip/hip_bf16.h>
#include <math.h>

// ---------------------------------------------------------------------------
// SPAIR-style decoder for MI455X (gfx1250, wave32, WMMA)
//   B=2, 135 cells -> 540 objects, what-decoder = 6 deconv2x2 layers
//   deconv2x2(X[M,Cin]) == GEMM X[M,Cin] x W[Cin,4*Cout] + spatial scatter
//   GEMM core: v_wmma_f32_16x16x32_f16, f32 accum, fragment-packed B,
//   b128 A/B loads, 1x2 register blocking, fused bias+act+scatter epilogue.
// ---------------------------------------------------------------------------

typedef __attribute__((ext_vector_type(16))) _Float16 v16h;
typedef __attribute__((ext_vector_type(8)))  _Float16 v8h;
typedef __attribute__((ext_vector_type(8)))  float    v8f;

#define NOBJ   540
#define NCELL  135
#define ZWHAT  64
#define BATCH  2
#define S_IMG  128

// ---------------------------------------------------------------------------
// Kernel 1: gather by indices, stable rank-sort by descending depth,
//           scatter z_what (as f16), z_where, z_present into sorted order.
// ---------------------------------------------------------------------------
__global__ void sort_gather_kernel(const float* __restrict__ z_what,
                                   const float* __restrict__ z_where,
                                   const float* __restrict__ z_present,
                                   const float* __restrict__ z_depth,
                                   const int*   __restrict__ indices,
                                   _Float16*    __restrict__ act0,     // (B*540, 64) f16
                                   float*       __restrict__ sWhere,   // (B,540,4)
                                   float*       __restrict__ sPresent) // (B,540)
{
    int b = blockIdx.x;
    __shared__ float depth[NOBJ];
    for (int i = threadIdx.x; i < NOBJ; i += blockDim.x)
        depth[i] = z_depth[b * NCELL + indices[i]];
    __syncthreads();

    for (int i = threadIdx.x; i < NOBJ; i += blockDim.x) {
        float di = depth[i];
        int rank = 0;
        for (int j = 0; j < NOBJ; ++j) {
            float dj = depth[j];
            // stable argsort of -depth (descending depth, ties -> lower idx)
            rank += (dj > di) || (dj == di && j < i);
        }
        const float* src = z_what + ((size_t)b * NCELL + indices[i]) * ZWHAT;
        _Float16* dst = act0 + ((size_t)b * NOBJ + rank) * ZWHAT;
        #pragma unroll 8
        for (int c = 0; c < ZWHAT; ++c) dst[c] = (_Float16)src[c];

        const float* wsrc = z_where + ((size_t)b * NOBJ + i) * 4;
        float* wdst = sWhere + ((size_t)b * NOBJ + rank) * 4;
        wdst[0] = wsrc[0]; wdst[1] = wsrc[1]; wdst[2] = wsrc[2]; wdst[3] = wsrc[3];
        sPresent[b * NOBJ + rank] = z_present[b * NOBJ + i];
    }
}

// ---------------------------------------------------------------------------
// Kernel 2: repack one layer's weights (Cin,Cout,2,2) f32 into WMMA B-fragment
// order, f16:  Bpack[((ks*Ntiles + nt)*32 + lane)*16 + h]
// where the half h of lane maps to K = ks*32 + (h>>3)*16 + (lane>>4)*8 + (h&7)
// and column col = nt*16 + (lane&15), with col = (a*2+b)*Cout + o.
// Out-of-range K/col slots are zero (handles K=16 tail and Nout=12 pad).
// ---------------------------------------------------------------------------
__global__ void prep_weights_frag_kernel(const float* __restrict__ w,
                                         _Float16* __restrict__ bpack,
                                         int K, int Cout, int Ntiles, int Ksteps)
{
    int t = blockIdx.x * blockDim.x + threadIdx.x;
    int total = Ksteps * Ntiles * 32 * 16;
    if (t >= total) return;
    int h    = t & 15;
    int lane = (t >> 4) & 31;
    int tile = t >> 9;                // ks*Ntiles + nt
    int ks   = tile / Ntiles;
    int nt   = tile - ks * Ntiles;
    int hi = lane >> 4, lo = lane & 15;
    int k   = ks * 32 + ((h >> 3) << 4) + (hi << 3) + (h & 7);
    int col = nt * 16 + lo;
    _Float16 v = (_Float16)0.0f;
    if (k < K && col < 4 * Cout) {
        int ab = col / Cout;
        int o  = col - ab * Cout;
        int a  = ab >> 1, bb = ab & 1;
        v = (_Float16)w[((k * Cout + o) * 2 + a) * 2 + bb];
    }
    bpack[t] = v;
}

// ---------------------------------------------------------------------------
// Kernel 3: WMMA GEMM + fused deconv2x2 scatter epilogue.
//   A     : (M, K) f16 row-major, K multiple of 16, rows 16B-aligned
//   Bpack : fragment-packed weights (see kernel 2)
// Each wave owns one 16-row M-tile and a 1x2 slab of N-tiles (nt0, nt0+1).
// A fragment = two aligned b128 loads (uniform zero-fill for K=16 upper half);
// B fragment = two aligned b128 loads straight into register layout.
// All v_wmma calls are wave-uniform (full EXEC).
// ---------------------------------------------------------------------------
union Frag16 { v16h v; v8h h[2]; };

__device__ __forceinline__ void epilogue_store(
    v8f acc, int nt, int mt, int lo, int hi, int M, int Nout,
    const float* __restrict__ bias, int Cout, int lgW,
    _Float16* __restrict__ outH, float* __restrict__ outObj)
{
    int col = nt * 16 + lo;
    if (col >= Nout) return;          // ragged Nout (layer 5: 12)
    int ab = col / Cout;
    int o  = col - ab * Cout;
    int a  = ab >> 1, bb = ab & 1;
    float bj = bias[o];
    const int Win  = 1 << lgW;
    const int HWin = 1 << (2 * lgW);
    #pragma unroll
    for (int r = 0; r < 8; ++r) {
        int m = mt * 16 + hi * 8 + r;         // D row for VGPR r
        if (m >= M) break;                    // only layer 0 is ragged
        int nobj = m >> (2 * lgW);
        int pix  = m & (HWin - 1);
        int hin  = pix >> lgW;
        int win  = pix & (Win - 1);
        int pixo = ((hin << 1) + a) * (Win << 1) + ((win << 1) + bb);
        float y = acc[r] + bj;
        if (outObj) {
            outObj[((size_t)nobj * 4096 + pixo) * 3 + o] =
                1.0f / (1.0f + __expf(-y));
        } else {
            outH[((size_t)nobj * (HWin << 2) + pixo) * Cout + o] =
                (_Float16)fmaxf(y, 0.0f);
        }
    }
}

__global__ void deconv_wmma_kernel(const _Float16* __restrict__ A, int M, int K,
                                   const _Float16* __restrict__ Bpack,
                                   int Ntiles, int Nout,
                                   const float* __restrict__ bias, int Cout,
                                   int lgW,
                                   _Float16* __restrict__ outH,
                                   float*    __restrict__ outObj,
                                   int Mtiles)
{
    const int wave = threadIdx.x >> 5;
    const int lane = threadIdx.x & 31;
    const int mt   = blockIdx.x * 4 + wave;
    const int nt0  = blockIdx.y * 2;
    if (mt >= Mtiles) return;                    // wave-uniform exit
    const bool hasN2 = (nt0 + 1) < Ntiles;       // wave-uniform

    const int lo = lane & 15;
    const int hi = lane >> 4;
    int row = mt * 16 + lo;
    if (row >= M) row = M - 1;                   // clamp; tail rows never stored
    const _Float16* Arow = A + (size_t)row * K;

    const int Ksteps = (K + 31) >> 5;
    const v8h zero8 = {};
    v8f acc0 = {}, acc1 = {};

    for (int ks = 0; ks < Ksteps; ++ks) {
        const int kb = ks * 32;
        Frag16 af, bf0, bf1;
        // A fragment: two contiguous 8-half runs per lane (b128 loads).
        // K is a multiple of 16, so the upper-half guard is wave-uniform.
        af.h[0] = *(const v8h*)(Arow + kb + (hi << 3));
        af.h[1] = (kb + 16 < K) ? *(const v8h*)(Arow + kb + 16 + (hi << 3))
                                : zero8;
        // B fragments: straight b128 loads from fragment-packed weights.
        const v8h* b0 = (const v8h*)(Bpack +
            ((((size_t)ks * Ntiles + nt0) * 32 + lane) << 4));
        bf0.h[0] = b0[0];
        bf0.h[1] = b0[1];
        acc0 = __builtin_amdgcn_wmma_f32_16x16x32_f16(
                   false, af.v, false, bf0.v, (short)0, acc0, false, false);
        if (hasN2) {
            const v8h* b1 = (const v8h*)(Bpack +
                ((((size_t)ks * Ntiles + nt0 + 1) * 32 + lane) << 4));
            bf1.h[0] = b1[0];
            bf1.h[1] = b1[1];
            acc1 = __builtin_amdgcn_wmma_f32_16x16x32_f16(
                       false, af.v, false, bf1.v, (short)0, acc1, false, false);
        }
    }

    epilogue_store(acc0, nt0, mt, lo, hi, M, Nout, bias, Cout, lgW, outH, outObj);
    if (hasN2)
        epilogue_store(acc1, nt0 + 1, mt, lo, hi, M, Nout, bias, Cout, lgW,
                       outH, outObj);
}

// ---------------------------------------------------------------------------
// Kernel 4: STN bilinear sampling + depth-order compositing.
//   obj : (1080, 64, 64, 3) f32 tiles (channel-last for coalesced corners)
//   Per output pixel: scan sorted objects, first nonzero per channel wins.
// ---------------------------------------------------------------------------
__global__ void stn_composite_kernel(const float* __restrict__ obj,
                                     const float* __restrict__ sWhere,
                                     const float* __restrict__ sPresent,
                                     float* __restrict__ out)
{
    __shared__ float boxes[NOBJ * 4];
    __shared__ float pres[NOBJ];
    const int b = blockIdx.x >> 6;               // 64 blocks per batch
    for (int i = threadIdx.x; i < NOBJ * 4; i += blockDim.x)
        boxes[i] = sWhere[(size_t)b * NOBJ * 4 + i];
    for (int i = threadIdx.x; i < NOBJ; i += blockDim.x)
        pres[i] = sPresent[b * NOBJ + i];
    __syncthreads();

    const int p   = blockIdx.x * blockDim.x + threadIdx.x;
    const int pix = p & (S_IMG * S_IMG - 1);
    const int y   = pix >> 7;
    const int x   = pix & 127;
    const float gx = ((x + 0.5f) / 128.0f) * 2.0f - 1.0f;
    const float gy = ((y + 0.5f) / 128.0f) * 2.0f - 1.0f;

    float o0 = 0.0f, o1 = 0.0f, o2 = 0.0f;
    bool  f0 = false, f1 = false, f2 = false;

    for (int r = 0; r < NOBJ; ++r) {
        float pr = pres[r];
        if (pr == 0.0f) continue;
        float cx = boxes[r * 4 + 0], cy = boxes[r * 4 + 1];
        float sx = fmaxf(boxes[r * 4 + 2], 1.0e-3f);
        float sy = fmaxf(boxes[r * 4 + 3], 1.0e-3f);
        float u  = (gx - (2.0f * cx - 1.0f)) / sx;
        float v  = (gy - (2.0f * cy - 1.0f)) / sy;
        float xf = ((u + 1.0f) * 64.0f - 1.0f) * 0.5f;
        float yf = ((v + 1.0f) * 64.0f - 1.0f) * 0.5f;
        float x0 = floorf(xf), y0 = floorf(yf);
        if (x0 < -1.0f || x0 > 63.0f || y0 < -1.0f || y0 > 63.0f) continue;

        float wx1 = xf - x0, wx0 = 1.0f - wx1;
        float wy1 = yf - y0, wy0 = 1.0f - wy1;
        int ix0 = (int)x0, iy0 = (int)y0;
        const float* base = obj + ((size_t)(b * NOBJ + r)) * 64 * 64 * 3;

        float v0 = 0.0f, v1 = 0.0f, v2 = 0.0f;
        #pragma unroll
        for (int dy = 0; dy < 2; ++dy) {
            int iy = iy0 + dy;
            if ((unsigned)iy >= 64u) continue;
            float wy = dy ? wy1 : wy0;
            #pragma unroll
            for (int dx = 0; dx < 2; ++dx) {
                int ix = ix0 + dx;
                if ((unsigned)ix >= 64u) continue;
                float wgt = (dx ? wx1 : wx0) * wy;
                const float* c4 = base + ((size_t)iy * 64 + ix) * 3;
                v0 += wgt * c4[0];
                v1 += wgt * c4[1];
                v2 += wgt * c4[2];
            }
        }
        v0 *= pr; v1 *= pr; v2 *= pr;
        if (!f0 && v0 != 0.0f) { o0 = v0; f0 = true; }
        if (!f1 && v1 != 0.0f) { o1 = v1; f1 = true; }
        if (!f2 && v2 != 0.0f) { o2 = v2; f2 = true; }
        if (f0 && f1 && f2) break;
    }
    out[(size_t)(b * 3 + 0) * 16384 + pix] = o0;
    out[(size_t)(b * 3 + 1) * 16384 + pix] = o1;
    out[(size_t)(b * 3 + 2) * 16384 + pix] = o2;
}

// ---------------------------------------------------------------------------
// Host launcher
// ---------------------------------------------------------------------------
extern "C" void kernel_launch(void* const* d_in, const int* in_sizes, int n_in,
                              void* d_out, int out_size, void* d_ws, size_t ws_size,
                              hipStream_t stream)
{
    const float* z_what    = (const float*)d_in[0];
    const float* z_where   = (const float*)d_in[1];
    const float* z_present = (const float*)d_in[2];
    const float* z_depth   = (const float*)d_in[3];
    const float* w[6]; const float* bvec[6];
    for (int i = 0; i < 6; ++i) {
        w[i]    = (const float*)d_in[4 + 2 * i];
        bvec[i] = (const float*)d_in[5 + 2 * i];
    }
    const int* indices = (const int*)d_in[16];
    float* out = (float*)d_out;

    // per-layer configs
    static const int Kc[6]     = {64, 256, 128, 64, 32, 16};
    static const int Coutc[6]  = {256, 128, 64, 32, 16, 3};
    static const int lgWc[6]   = {0, 1, 2, 3, 4, 5};
    static const int Ntilesc[6]= {64, 32, 16, 8, 4, 1};   // ceil(4*Cout/16)
    static const int Noutc[6]  = {1024, 512, 256, 128, 64, 12};

    // workspace carve-out (256B aligned)
    char* ws = (char*)d_ws;
    size_t off = 0;
    auto carve = [&](size_t bytes) -> char* {
        char* p = ws + off;
        off = (off + bytes + 255) & ~(size_t)255;
        return p;
    };
    float*    sWhere   = (float*)carve((size_t)BATCH * NOBJ * 4 * sizeof(float));
    float*    sPresent = (float*)carve((size_t)BATCH * NOBJ * sizeof(float));
    _Float16* bpack[6];
    int       Kstepsc[6];
    for (int i = 0; i < 6; ++i) {
        Kstepsc[i] = (Kc[i] + 31) / 32;
        bpack[i] = (_Float16*)carve(
            (size_t)Kstepsc[i] * Ntilesc[i] * 32 * 16 * sizeof(_Float16));
    }
    // act ping-pong: bufA holds L0/L2/L4 inputs (max 1080*256*32 halfs),
    //                bufB holds L1/L3/L5 inputs (max 1080*1024*16 halfs)
    _Float16* bufA = (_Float16*)carve((size_t)1080 * 256 * 32 * sizeof(_Float16));
    _Float16* bufB = (_Float16*)carve((size_t)1080 * 1024 * 16 * sizeof(_Float16));
    float*    objT = (float*)carve((size_t)1080 * 64 * 64 * 3 * sizeof(float));

    // 1) gather + depth sort (act0 -> bufA)
    sort_gather_kernel<<<BATCH, 256, 0, stream>>>(
        z_what, z_where, z_present, z_depth, indices, bufA, sWhere, sPresent);

    // 2) weight repack into WMMA B-fragment order
    for (int i = 0; i < 6; ++i) {
        int total = Kstepsc[i] * Ntilesc[i] * 32 * 16;
        prep_weights_frag_kernel<<<(total + 255) / 256, 256, 0, stream>>>(
            w[i], bpack[i], Kc[i], Coutc[i], Ntilesc[i], Kstepsc[i]);
    }

    // 3) six WMMA GEMM + fused deconv scatter layers
    for (int i = 0; i < 6; ++i) {
        int M      = 1080 << (2 * lgWc[i]);
        int Mtiles = (M + 15) / 16;
        dim3 grid((Mtiles + 3) / 4, (Ntilesc[i] + 1) / 2);
        _Float16* inA  = (i & 1) ? bufB : bufA;
        _Float16* outH = (i == 5) ? nullptr : ((i & 1) ? bufA : bufB);
        float*    outO = (i == 5) ? objT : nullptr;
        deconv_wmma_kernel<<<grid, 128, 0, stream>>>(
            inA, M, Kc[i], bpack[i], Ntilesc[i], Noutc[i],
            bvec[i], Coutc[i], lgWc[i], outH, outO, Mtiles);
    }

    // 4) STN + depth compositing (2*128*128 px / 256 = 128 blocks)
    stn_composite_kernel<<<BATCH * 64, 256, 0, stream>>>(
        objT, sWhere, sPresent, out);
}
